// BinarizedRNN_29472065585438
// MI455X (gfx1250) — matrix operand order repaired and checked by reading, not compiled, not run on hardware
//
#include <hip/hip_runtime.h>
#include <hip/hip_bf16.h>
#include <cstdint>

// ---------------------------------------------------------------------------
// BinarizedRNN on gfx1250 (MI455X), wave32 + v_wmma_f32_16x16x32_f16.
//   T=512, B=256, I=256, H=512, O=256.
// Per step t:  pre = x_t @ sign(W_in)^T + g_t * (h @ sign(W_h)^T)
//              c   = clip(pre,-1,1)
//              hn  = (c*sig(10c)*l1 + c*sig(-10c)*l2) / sqrt(var_B(c)+eps)
//              out = hn @ sign(W_out)^T
//
// If ws_size permits, xin = x @ sign(W_in)^T is precomputed for all T in one
// wide WMMA GEMM (off the serial critical path); otherwise it is fused into
// each step. All GEMM inner loops are double-buffered so B-fragment loads for
// kt+1 overlap the WMMAs of kt.
// ---------------------------------------------------------------------------

#define Tt 512
#define Bb 256
#define Ii 256
#define Hh 512
#define Oo 256

typedef __attribute__((ext_vector_type(16))) _Float16 v16h;
typedef __attribute__((ext_vector_type(8)))  _Float16 v8h;
typedef __attribute__((ext_vector_type(8)))  float    v8f;

__device__ __forceinline__ v16h cat8(v8h lo, v8h hi) {
  return __builtin_shufflevector(lo, hi, 0,1,2,3,4,5,6,7,8,9,10,11,12,13,14,15);
}

__device__ __forceinline__ v8f wmma_f16(v16h a, v16h b, v8f c) {
  return __builtin_amdgcn_wmma_f32_16x16x32_f16(
      false, a, false, b, (short)0, c, false, false);
}

// B-fragment load: per-lane-contiguous 16 halves (32B) per fragment.
__device__ __forceinline__ v16h loadB(const _Float16* __restrict__ F, int NT,
                                      int kt, int nt, int lane) {
  const _Float16* fb = F + (size_t)(kt * NT + nt) * 512 + lane * 16;
  return cat8(*(const v8h*)fb, *(const v8h*)(fb + 8));
}

// A-fragment (16-bit 16x32) from row-major f16: lanes<16 take K 0-7/16-23,
// lanes>=16 take K 8-15/24-31 of the 32-wide K block.
__device__ __forceinline__ v16h loadA_h(const _Float16* __restrict__ row,
                                        int kb) {
  return cat8(*(const v8h*)(row + kb), *(const v8h*)(row + kb + 16));
}

// A-fragment from row-major f32 with inline f32->f16 conversion.
__device__ __forceinline__ v16h loadA_f32(const float* __restrict__ row,
                                          int kb) {
  float4 f0 = *(const float4*)(row + kb);
  float4 f1 = *(const float4*)(row + kb + 4);
  float4 f2 = *(const float4*)(row + kb + 16);
  float4 f3 = *(const float4*)(row + kb + 20);
  v16h a;
  a[0]=(_Float16)f0.x;  a[1]=(_Float16)f0.y;  a[2]=(_Float16)f0.z;  a[3]=(_Float16)f0.w;
  a[4]=(_Float16)f1.x;  a[5]=(_Float16)f1.y;  a[6]=(_Float16)f1.z;  a[7]=(_Float16)f1.w;
  a[8]=(_Float16)f2.x;  a[9]=(_Float16)f2.y;  a[10]=(_Float16)f2.z; a[11]=(_Float16)f2.w;
  a[12]=(_Float16)f3.x; a[13]=(_Float16)f3.y; a[14]=(_Float16)f3.z; a[15]=(_Float16)f3.w;
  return a;
}

// ---------------------------------------------------------------------------
// Pack sign(W) into B-matrix fragments (16-bit B 32x16 layout):
//   element j (0..15): lanes 0-15 -> K = (j&7) + (j&8 ? 16 : 0); lanes 16-31 -> that + 8;
//   N = lane%16.  Storage: frag[(kt*NT+nt)*512 + lane*16 + j].
// ---------------------------------------------------------------------------
__global__ void __launch_bounds__(256)
pack_weights(const float* __restrict__ W_in, const float* __restrict__ W_h,
             const float* __restrict__ W_out,
             _Float16* __restrict__ F_in, _Float16* __restrict__ F_h,
             _Float16* __restrict__ F_out) {
  int fid = blockIdx.x;
  const float* W; _Float16* F; int K, NT, local;
  if (fid < 256)      { W = W_in;  F = F_in;  K = Ii; NT = 32; local = fid; }
  else if (fid < 768) { W = W_h;   F = F_h;   K = Hh; NT = 32; local = fid - 256; }
  else                { W = W_out; F = F_out; K = Hh; NT = 16; local = fid - 768; }
  int kt = local / NT, nt = local % NT;
  for (int e = threadIdx.x; e < 512; e += 256) {
    int lane = e >> 4, j = e & 15;
    int v = j >> 1, hh = j & 1;
    int nl = lane & 15;
    int kl = ((lane & 16) ? 8 : 0) + ((2 * v + hh) & 7) + (((2 * v + hh) & 8) ? 16 : 0);
    float w = W[(size_t)(nt * 16 + nl) * K + (kt * 32 + kl)];
    float s = (w > 0.f) ? 1.f : ((w < 0.f) ? -1.f : 0.f);
    F[(size_t)local * 512 + e] = (_Float16)s;
  }
}

__global__ void __launch_bounds__(256)
zero_h(uint32_t* __restrict__ p) {
  p[blockIdx.x * 256 + threadIdx.x] = 0u;   // 65536 dwords = 131072 halves
}

// ---------------------------------------------------------------------------
// Wide precompute: xin[T*B, H] = x[T*B, I] @ sign(W_in)^T, f16 output.
// grid = T*B/16 = 8192 strips; 8 waves x 4 n-tiles cover H=512.
// ---------------------------------------------------------------------------
__global__ void __launch_bounds__(256)
xin_gemm(const float* __restrict__ x, const _Float16* __restrict__ Fin,
         _Float16* __restrict__ xin) {
  const int strip = blockIdx.x;
  const int wave = threadIdx.x >> 5;
  const int lane = threadIdx.x & 31;
  const int half16 = (lane & 16) ? 1 : 0;
  const int mrow = strip * 16 + (lane & 15);

  v8f acc[4];
  #pragma unroll
  for (int q = 0; q < 4; ++q) acc[q] = (v8f)(0.f);

  const float* xrow = x + (size_t)mrow * Ii;
  v16h a_cur = loadA_f32(xrow, half16 * 8);
  v16h b_cur[4], b_nxt[4];
  #pragma unroll
  for (int q = 0; q < 4; ++q) b_cur[q] = loadB(Fin, 32, 0, wave * 4 + q, lane);

  #pragma unroll
  for (int kt = 0; kt < 8; ++kt) {
    v16h a_nxt = a_cur;
    #pragma unroll
    for (int q = 0; q < 4; ++q) b_nxt[q] = b_cur[q];
    if (kt < 7) {
      a_nxt = loadA_f32(xrow, (kt + 1) * 32 + half16 * 8);
      #pragma unroll
      for (int q = 0; q < 4; ++q)
        b_nxt[q] = loadB(Fin, 32, kt + 1, wave * 4 + q, lane);
    }
    #pragma unroll
    for (int q = 0; q < 4; ++q) acc[q] = wmma_f16(a_cur, b_cur[q], acc[q]);
    a_cur = a_nxt;
    #pragma unroll
    for (int q = 0; q < 4; ++q) b_cur[q] = b_nxt[q];
  }

  #pragma unroll
  for (int q = 0; q < 4; ++q) {
    int n = (wave * 4 + q) * 16 + (lane & 15);
    #pragma unroll
    for (int r = 0; r < 8; ++r) {
      int m = strip * 16 + r + half16 * 8;
      xin[(size_t)m * Hh + n] = (_Float16)acc[q][r];
    }
  }
}

// ---------------------------------------------------------------------------
// Common epilogue for the step kernels: clip, store pre-activations,
// per-strip column sum / sumsq (batch-axis partial reduction).
// ---------------------------------------------------------------------------
__device__ __forceinline__ void step_epilogue(
    v8f* acc, int strip, int wave, int lane, int half16,
    float* __restrict__ pre_clip, float* __restrict__ psum,
    float* __restrict__ psq) {
  #pragma unroll
  for (int q = 0; q < 4; ++q) {
    int n = (wave * 4 + q) * 16 + (lane & 15);
    float s = 0.f, sq = 0.f;
    #pragma unroll
    for (int r = 0; r < 8; ++r) {
      float c = acc[q][r];
      c = fminf(1.f, fmaxf(-1.f, c));
      int m = strip * 16 + r + half16 * 8;
      pre_clip[(size_t)m * Hh + n] = c;
      s += c; sq += c * c;
    }
    s  += __shfl_xor(s, 16, 32);
    sq += __shfl_xor(sq, 16, 32);
    if (!half16) {
      psum[strip * Hh + n] = s;
      psq [strip * Hh + n] = sq;
    }
  }
}

// ---------------------------------------------------------------------------
// Step kernel, precomputed-xin path: acc = h @ Wb_h^T; acc = g*acc + xin_t.
// grid = 16 strips, 256 threads = 8 waves x 4 n-tiles.
// ---------------------------------------------------------------------------
__global__ void __launch_bounds__(256)
rnn_step_pre(const _Float16* __restrict__ xin_t, const _Float16* __restrict__ hbuf,
             const _Float16* __restrict__ Fh, const float* __restrict__ gates,
             int t, float* __restrict__ pre_clip, float* __restrict__ psum,
             float* __restrict__ psq) {
  const int strip = blockIdx.x;
  const int wave = threadIdx.x >> 5;
  const int lane = threadIdx.x & 31;
  const int half16 = (lane & 16) ? 1 : 0;
  const int mrow = strip * 16 + (lane & 15);

  v8f acc[4];
  #pragma unroll
  for (int q = 0; q < 4; ++q) acc[q] = (v8f)(0.f);

  const _Float16* hrow = hbuf + (size_t)mrow * Hh;
  v16h a_cur = loadA_h(hrow, half16 * 8);
  v16h b_cur[4], b_nxt[4];
  #pragma unroll
  for (int q = 0; q < 4; ++q) b_cur[q] = loadB(Fh, 32, 0, wave * 4 + q, lane);

  #pragma unroll
  for (int kt = 0; kt < 16; ++kt) {
    v16h a_nxt = a_cur;
    #pragma unroll
    for (int q = 0; q < 4; ++q) b_nxt[q] = b_cur[q];
    if (kt < 15) {
      a_nxt = loadA_h(hrow, (kt + 1) * 32 + half16 * 8);
      #pragma unroll
      for (int q = 0; q < 4; ++q)
        b_nxt[q] = loadB(Fh, 32, kt + 1, wave * 4 + q, lane);
    }
    #pragma unroll
    for (int q = 0; q < 4; ++q) acc[q] = wmma_f16(a_cur, b_cur[q], acc[q]);
    a_cur = a_nxt;
    #pragma unroll
    for (int q = 0; q < 4; ++q) b_cur[q] = b_nxt[q];
  }

  float g = gates[t];
  #pragma unroll
  for (int q = 0; q < 4; ++q) {
    int n = (wave * 4 + q) * 16 + (lane & 15);
    #pragma unroll
    for (int r = 0; r < 8; ++r) {
      int m = strip * 16 + r + half16 * 8;
      acc[q][r] = acc[q][r] * g + (float)xin_t[(size_t)m * Hh + n];
    }
  }

  step_epilogue(acc, strip, wave, lane, half16, pre_clip, psum, psq);
}

// ---------------------------------------------------------------------------
// Step kernel, fused fallback (small workspace): x-projection done in-step.
// ---------------------------------------------------------------------------
__global__ void __launch_bounds__(256)
rnn_step_fused(const float* __restrict__ x_t, const _Float16* __restrict__ hbuf,
               const _Float16* __restrict__ Fh, const _Float16* __restrict__ Fin,
               const float* __restrict__ gates, int t,
               float* __restrict__ pre_clip, float* __restrict__ psum,
               float* __restrict__ psq) {
  const int strip = blockIdx.x;
  const int wave = threadIdx.x >> 5;
  const int lane = threadIdx.x & 31;
  const int half16 = (lane & 16) ? 1 : 0;
  const int mrow = strip * 16 + (lane & 15);

  v8f acc[4];
  #pragma unroll
  for (int q = 0; q < 4; ++q) acc[q] = (v8f)(0.f);

  // ---- h @ Wb_h^T, K = 512, double-buffered ----
  const _Float16* hrow = hbuf + (size_t)mrow * Hh;
  {
    v16h a_cur = loadA_h(hrow, half16 * 8);
    v16h b_cur[4], b_nxt[4];
    #pragma unroll
    for (int q = 0; q < 4; ++q) b_cur[q] = loadB(Fh, 32, 0, wave * 4 + q, lane);
    #pragma unroll
    for (int kt = 0; kt < 16; ++kt) {
      v16h a_nxt = a_cur;
      #pragma unroll
      for (int q = 0; q < 4; ++q) b_nxt[q] = b_cur[q];
      if (kt < 15) {
        a_nxt = loadA_h(hrow, (kt + 1) * 32 + half16 * 8);
        #pragma unroll
        for (int q = 0; q < 4; ++q)
          b_nxt[q] = loadB(Fh, 32, kt + 1, wave * 4 + q, lane);
      }
      #pragma unroll
      for (int q = 0; q < 4; ++q) acc[q] = wmma_f16(a_cur, b_cur[q], acc[q]);
      a_cur = a_nxt;
      #pragma unroll
      for (int q = 0; q < 4; ++q) b_cur[q] = b_nxt[q];
    }
  }

  float g = gates[t];
  #pragma unroll
  for (int q = 0; q < 4; ++q)
    #pragma unroll
    for (int r = 0; r < 8; ++r) acc[q][r] *= g;

  // ---- + x_t @ Wb_in^T, K = 256, double-buffered ----
  const float* xrow = x_t + (size_t)mrow * Ii;
  {
    v16h a_cur = loadA_f32(xrow, half16 * 8);
    v16h b_cur[4], b_nxt[4];
    #pragma unroll
    for (int q = 0; q < 4; ++q) b_cur[q] = loadB(Fin, 32, 0, wave * 4 + q, lane);
    #pragma unroll
    for (int kt = 0; kt < 8; ++kt) {
      v16h a_nxt = a_cur;
      #pragma unroll
      for (int q = 0; q < 4; ++q) b_nxt[q] = b_cur[q];
      if (kt < 7) {
        a_nxt = loadA_f32(xrow, (kt + 1) * 32 + half16 * 8);
        #pragma unroll
        for (int q = 0; q < 4; ++q)
          b_nxt[q] = loadB(Fin, 32, kt + 1, wave * 4 + q, lane);
      }
      #pragma unroll
      for (int q = 0; q < 4; ++q) acc[q] = wmma_f16(a_cur, b_cur[q], acc[q]);
      a_cur = a_nxt;
      #pragma unroll
      for (int q = 0; q < 4; ++q) b_cur[q] = b_nxt[q];
    }
  }

  step_epilogue(acc, strip, wave, lane, half16, pre_clip, psum, psq);
}

// ---------------------------------------------------------------------------
// Kernel O: finish batch-norm reduction, apply sign-sensitive norm, write
// next-step h (f16), compute out_t = hn @ Wb_out^T from LDS.
// ---------------------------------------------------------------------------
__global__ void __launch_bounds__(256)
rnn_norm_out(const float* __restrict__ pre_clip, const float* __restrict__ psum,
             const float* __restrict__ psq,
             const float* __restrict__ l1, const float* __restrict__ l2,
             const _Float16* __restrict__ Fout, _Float16* __restrict__ hbuf,
             float* __restrict__ out_t) {
  __shared__ float    s_inv[Hh];
  __shared__ float    s_l1[Hh];
  __shared__ float    s_l2[Hh];
  __shared__ _Float16 s_hn[16 * Hh];   // 16 KB

  const int strip = blockIdx.x;

  for (int f = threadIdx.x; f < Hh; f += 256) {
    float S = 0.f, Q = 0.f;
    #pragma unroll
    for (int s = 0; s < 16; ++s) { S += psum[s * Hh + f]; Q += psq[s * Hh + f]; }
    float mean = S * (1.f / 256.f);
    float var  = (Q - 256.f * mean * mean) * (1.f / 255.f);
    s_inv[f] = rsqrtf(var + 1e-5f);
    s_l1[f] = l1[f];
    s_l2[f] = l2[f];
  }
  __syncthreads();

  for (int e = threadIdx.x; e < 16 * Hh; e += 256) {
    int r = e >> 9, f = e & (Hh - 1);
    float c = pre_clip[(size_t)(strip * 16 + r) * Hh + f];
    float sig = 1.f / (1.f + __expf(-10.f * c));
    float hn = c * (sig * s_l1[f] + (1.f - sig) * s_l2[f]) * s_inv[f];
    _Float16 hh = (_Float16)hn;
    s_hn[e] = hh;
    hbuf[(size_t)(strip * 16 + r) * Hh + f] = hh;
  }
  __syncthreads();

  const int wave = threadIdx.x >> 5;
  const int lane = threadIdx.x & 31;
  const int half16 = (lane & 16) ? 1 : 0;
  const int arow = lane & 15;

  v8f acc[2];
  #pragma unroll
  for (int q = 0; q < 2; ++q) acc[q] = (v8f)(0.f);

  const _Float16* hp = s_hn + arow * Hh;
  v16h a_cur = loadA_h(hp, half16 * 8);
  v16h b_cur[2], b_nxt[2];
  #pragma unroll
  for (int q = 0; q < 2; ++q) b_cur[q] = loadB(Fout, 16, 0, wave * 2 + q, lane);

  #pragma unroll
  for (int kt = 0; kt < 16; ++kt) {
    v16h a_nxt = a_cur;
    #pragma unroll
    for (int q = 0; q < 2; ++q) b_nxt[q] = b_cur[q];
    if (kt < 15) {
      a_nxt = loadA_h(hp, (kt + 1) * 32 + half16 * 8);
      #pragma unroll
      for (int q = 0; q < 2; ++q)
        b_nxt[q] = loadB(Fout, 16, kt + 1, wave * 2 + q, lane);
    }
    #pragma unroll
    for (int q = 0; q < 2; ++q) acc[q] = wmma_f16(a_cur, b_cur[q], acc[q]);
    a_cur = a_nxt;
    #pragma unroll
    for (int q = 0; q < 2; ++q) b_cur[q] = b_nxt[q];
  }

  #pragma unroll
  for (int q = 0; q < 2; ++q) {
    int n = (wave * 2 + q) * 16 + (lane & 15);
    #pragma unroll
    for (int r = 0; r < 8; ++r) {
      int m = strip * 16 + r + half16 * 8;
      out_t[(size_t)m * Oo + n] = acc[q][r];
    }
  }
}

// ---------------------------------------------------------------------------
extern "C" void kernel_launch(void* const* d_in, const int* in_sizes, int n_in,
                              void* d_out, int out_size, void* d_ws, size_t ws_size,
                              hipStream_t stream) {
  (void)in_sizes; (void)n_in; (void)out_size;
  const float* x     = (const float*)d_in[0];   // [T,B,I]
  const float* W_in  = (const float*)d_in[1];   // [H,I]
  const float* W_h   = (const float*)d_in[2];   // [H,H]
  const float* W_out = (const float*)d_in[3];   // [O,H]
  const float* gates = (const float*)d_in[4];   // [T]
  const float* l1    = (const float*)d_in[5];   // [H]
  const float* l2    = (const float*)d_in[6];   // [H]
  float* out = (float*)d_out;                   // [T,B,O]

  // workspace carve (~1.9 MB base, L2-resident for the whole scan)
  _Float16* F_in  = (_Float16*)d_ws;            // 8*32*512   = 131072 halves
  _Float16* F_h   = F_in  + 131072;             // 16*32*512  = 262144
  _Float16* F_out = F_h   + 262144;             // 16*16*512  = 131072
  _Float16* hbuf  = F_out + 131072;             // B*H        = 131072
  float* pre  = (float*)(hbuf + 131072);        // B*H floats = 131072
  float* psum = pre  + (size_t)Bb * Hh;         // 16*H = 8192
  float* psq  = psum + 16 * Hh;                 // 16*H = 8192
  _Float16* xin = (_Float16*)(psq + 16 * Hh);   // optional: T*B*H halves

  const size_t base_bytes = (size_t)((char*)xin - (char*)d_ws);
  const size_t xin_bytes  = (size_t)Tt * Bb * Hh * sizeof(_Float16);
  const bool precompute = ws_size >= base_bytes + xin_bytes;

  pack_weights<<<1024, 256, 0, stream>>>(W_in, W_h, W_out, F_in, F_h, F_out);
  zero_h<<<256, 256, 0, stream>>>((uint32_t*)hbuf);

  if (precompute) {
    xin_gemm<<<(Tt * Bb) / 16, 256, 0, stream>>>(x, F_in, xin);
    for (int t = 0; t < Tt; ++t) {
      rnn_step_pre<<<16, 256, 0, stream>>>(xin + (size_t)t * Bb * Hh, hbuf,
                                           F_h, gates, t, pre, psum, psq);
      rnn_norm_out<<<16, 256, 0, stream>>>(pre, psum, psq, l1, l2, F_out, hbuf,
                                           out + (size_t)t * Bb * Oo);
    }
  } else {
    for (int t = 0; t < Tt; ++t) {
      rnn_step_fused<<<16, 256, 0, stream>>>(x + (size_t)t * Bb * Ii, hbuf,
                                             F_h, F_in, gates, t, pre, psum, psq);
      rnn_norm_out<<<16, 256, 0, stream>>>(pre, psum, psq, l1, l2, F_out, hbuf,
                                           out + (size_t)t * Bb * Oo);
    }
  }
}
